// NTXentLoss_91199335563999
// MI455X (gfx1250) — matrix operand order repaired and checked by reading, not compiled, not run on hardware
//
#include <hip/hip_runtime.h>

typedef __attribute__((ext_vector_type(16))) __bf16 v16bf;
typedef __attribute__((ext_vector_type(8)))  float  v8f;
typedef unsigned short u16;
typedef unsigned int   u32;

#define TWO_N   8192
#define N_HALF  4096
#define DIMK    256
#define CT      64        // columns per LDS tile
#define LSTRIDE 272       // LDS row stride in u16 (256 + 16 pad)
#define NBLK    (TWO_N / CT)

// 2*log2(e): exp(2*acc - 2) == exp2(acc*C - C), maps to a single v_exp_f32
#define EXP2C   2.8853900817779268f
#define LN2     0.6931471805599453f

union FragBF {
  v16bf v;
  uint4 q[2];
};

union Pack8 {
  uint4 q;
  u16   h[8];
};

// float -> bf16 round-to-nearest-even (raw bits)
__device__ __forceinline__ u16 f2bf(float f) {
  union { float f; u32 u; } c;
  c.f = f;
  u32 r = 0x7FFFu + ((c.u >> 16) & 1u);
  return (u16)((c.u + r) >> 16);
}

__device__ __forceinline__ float bf2f(u16 h) {
  union { u32 u; float f; } c;
  c.u = ((u32)h) << 16;
  return c.f;
}

// -------- Kernel 1: L2-normalize rows of concat(zjs, zis) into bf16 R-hat ----
__global__ __launch_bounds__(256) void nk_normalize(const float* __restrict__ zis,
                                                    const float* __restrict__ zjs,
                                                    u16* __restrict__ rhat) {
  const int row  = (blockIdx.x * blockDim.x + threadIdx.x) >> 5;  // one wave/row
  const int lane = threadIdx.x & 31;
  const float* src = (row < N_HALF) ? (zjs + (size_t)row * DIMK)
                                    : (zis + (size_t)(row - N_HALF) * DIMK);
  const float4* s4 = (const float4*)src;
  float4 a = s4[lane * 2 + 0];
  float4 b = s4[lane * 2 + 1];
  float ss = a.x * a.x + a.y * a.y + a.z * a.z + a.w * a.w +
             b.x * b.x + b.y * b.y + b.z * b.z + b.w * b.w;
#pragma unroll
  for (int d = 16; d > 0; d >>= 1) ss += __shfl_xor(ss, d, 32);
  const float inv = rsqrtf(fmaxf(ss, 1e-24f));
  Pack8 p;
  p.h[0] = f2bf(a.x * inv); p.h[1] = f2bf(a.y * inv);
  p.h[2] = f2bf(a.z * inv); p.h[3] = f2bf(a.w * inv);
  p.h[4] = f2bf(b.x * inv); p.h[5] = f2bf(b.y * inv);
  p.h[6] = f2bf(b.z * inv); p.h[7] = f2bf(b.w * inv);
  ((uint4*)(rhat + (size_t)row * DIMK))[lane] = p.q;
}

// -------- Kernel 1b: positive logits pos_i = 2 * dot(r_i, r_{(i+N)%2N}) -----
__global__ __launch_bounds__(256) void nk_pos(const u16* __restrict__ R,
                                              float* __restrict__ posv) {
  const int row  = (blockIdx.x * blockDim.x + threadIdx.x) >> 5;  // one wave/row
  const int lane = threadIdx.x & 31;
  const int prow = (row + N_HALF) & (TWO_N - 1);
  Pack8 a, b;
  a.q = ((const uint4*)(R + (size_t)row  * DIMK))[lane];
  b.q = ((const uint4*)(R + (size_t)prow * DIMK))[lane];
  float d = 0.0f;
#pragma unroll
  for (int k = 0; k < 8; ++k) d = fmaf(bf2f(a.h[k]), bf2f(b.h[k]), d);
#pragma unroll
  for (int s = 16; s > 0; s >>= 1) d += __shfl_xor(d, s, 32);
  if (lane == 0) posv[row] = 2.0f * d;   // logit = sim / TEMP
}

// Async global->LDS stage of one CT x 256 bf16 tile (no VGPR round trip;
// tracked with ASYNCcnt). ldsBase = wave-relative byte address of dest buffer.
__device__ __forceinline__ void load_tile_async(const u16* __restrict__ R,
                                                int n0, u32 ldsBase, int tid) {
#pragma unroll
  for (int t = 0; t < (CT * 32) / 256; ++t) {
    int idx = tid + t * 256;           // CT rows * 32 uint4 per row
    int row = idx >> 5;
    int c   = idx & 31;
    const u16* g = R + (size_t)(n0 + row) * DIMK + (size_t)c * 8;
    u32 lds = ldsBase + (u32)(row * LSTRIDE + c * 8) * 2u;
    asm volatile("global_load_async_to_lds_b128 %0, %1, off"
                 :: "v"(lds), "v"(g)
                 : "memory");
  }
}

// -------- Kernel 2: fused S = R R^T (bf16 WMMA) + fixed-max logsumexp -------
// Since |cos sim| <= 1, logits <= 2.0 = 1/T; fixed max M = 2.0:
//   lse_i = 2 + ln( sum_{j != i} exp(2*acc - 2) ),  every term in [e^-4, 1].
__global__ __launch_bounds__(256) void nk_main(const u16* __restrict__ R,
                                               float* __restrict__ rowLse) {
  __shared__ __align__(16) u16 tileB[2][CT * LSTRIDE];   // 2 x 34,816 B

  const int tid   = threadIdx.x;
  const int lane  = tid & 31;
  const int wv    = tid >> 5;          // 8 waves, 16 rows each
  const int mrow  = lane & 15;
  const int half  = lane >> 4;
  const int iBase = blockIdx.x * 128 + wv * 16;
  const int gRow  = iBase + mrow;

  // Preload A fragments for all 8 K-chunks (16x32 bf16 each).
  FragBF aF[8];
  const uint4* rp = (const uint4*)(R + (size_t)gRow * DIMK);  // 32 uint4 per row
#pragma unroll
  for (int kc = 0; kc < 8; ++kc) {
    aF[kc].q[0] = rp[kc * 4 + half];       // K = kc*32 + half*8 .. +7
    aF[kc].q[1] = rp[kc * 4 + half + 2];   // K = kc*32 + 16 + half*8 .. +7
  }

  float sm[8];
#pragma unroll
  for (int r = 0; r < 8; ++r) sm[r] = 0.0f;

  // LDS byte offset of the double buffer (flat-address low 32 bits == LDS addr).
  const u32 ldsBase = (u32)(uintptr_t)(&tileB[0][0]);

  load_tile_async(R, 0, ldsBase, tid);                 // prime the pipeline

  for (int blk = 0; blk < NBLK; ++blk) {
    const int n0 = blk * CT;
    asm volatile("s_wait_asynccnt 0x0" ::: "memory");  // own tile landed
    __syncthreads();                                   // everyone's landed
    if (blk + 1 < NBLK)                                // overlap next tile load
      load_tile_async(R, n0 + CT,
                      ldsBase + (u32)(((blk + 1) & 1) * (CT * LSTRIDE * 2)), tid);

    const u16* tb = &tileB[blk & 1][0];
#pragma unroll
    for (int nt = 0; nt < CT / 16; ++nt) {
      const u16* bb = tb + (nt * 16 + mrow) * LSTRIDE + half * 8;
      FragBF bF[8];
#pragma unroll
      for (int kc = 0; kc < 8; ++kc) {
        bF[kc].q[0] = *(const uint4*)(bb + kc * 32);
        bF[kc].q[1] = *(const uint4*)(bb + kc * 32 + 16);
      }
      v8f acc = {};
#pragma unroll
      for (int kc = 0; kc < 8; ++kc) {
        acc = __builtin_amdgcn_wmma_f32_16x16x32_bf16(
            false, aF[kc].v, false, bF[kc].v, (short)0, acc, false, false);
      }
      const int j = n0 + nt * 16 + mrow;   // global column for this lane
#pragma unroll
      for (int r = 0; r < 8; ++r) {
        // exp2(acc*C - C) == exp(2*acc - 2); branchless diagonal mask.
        float e = exp2f(fmaf(acc[r], EXP2C, -EXP2C));
        sm[r] += (j == (iBase + r + 8 * half)) ? 0.0f : e;
      }
    }
  }

  // Sum partial sums over the 16 lanes of each half; emit lse per row.
#pragma unroll
  for (int r = 0; r < 8; ++r) {
    float s = sm[r];
#pragma unroll
    for (int d = 1; d < 16; d <<= 1) s += __shfl_xor(s, d, 32);
    if (mrow == 0)
      rowLse[iBase + r + 8 * half] = fmaf(__log2f(s), LN2, 2.0f);  // 2 + ln(s)
  }
}

// -------- Kernel 3: loss = sum(lse_i - pos_i) / 2N --------------------------
__global__ __launch_bounds__(256) void nk_reduce(const float* __restrict__ rowLse,
                                                 const float* __restrict__ posv,
                                                 float* __restrict__ out) {
  __shared__ float buf[256];
  float a = 0.0f;
  for (int i = threadIdx.x; i < TWO_N; i += 256) a += rowLse[i] - posv[i];
  buf[threadIdx.x] = a;
  __syncthreads();
  for (int step = 128; step > 0; step >>= 1) {
    if (threadIdx.x < step) buf[threadIdx.x] += buf[threadIdx.x + step];
    __syncthreads();
  }
  if (threadIdx.x == 0) out[0] = buf[0] / (float)TWO_N;
}

extern "C" void kernel_launch(void* const* d_in, const int* in_sizes, int n_in,
                              void* d_out, int out_size, void* d_ws, size_t ws_size,
                              hipStream_t stream) {
  (void)in_sizes; (void)n_in; (void)out_size; (void)ws_size;
  const float* zis = (const float*)d_in[0];
  const float* zjs = (const float*)d_in[1];
  u16*   rhat   = (u16*)d_ws;                                        // 4 MB
  float* rowLse = (float*)((char*)d_ws + (size_t)TWO_N * DIMK * 2);  // 32 KB
  float* posv   = rowLse + TWO_N;                                    // 32 KB
  float* out    = (float*)d_out;

  nk_normalize<<<TWO_N / 8, 256, 0, stream>>>(zis, zjs, rhat);
  nk_pos<<<TWO_N / 8, 256, 0, stream>>>(rhat, posv);
  nk_main<<<TWO_N / 128, 256, 0, stream>>>(rhat, rowLse);
  nk_reduce<<<1, 256, 0, stream>>>(rowLse, posv, out);
}
